// FBOnlyKnownEyeMovementImageLoss_26448408609541
// MI455X (gfx1250) — matrix operand order repaired and checked by reading, not compiled, not run on hardware
//
#include <hip/hip_runtime.h>
#include <math.h>

// ---------------- problem constants ----------------
#define BATCH   8
#define IMG_H   160
#define IMG_W   256
#define P_TOT   40960          // H*W
#define CDIM    700
#define KTAPS   250
#define NBINS   750
#define NHIST   30
#define NJIT    60
#define NFRAMES 90
#define NCOLS   480            // B*NJIT
#define NPAD    512            // padded GEMM N
#define MPAD    704            // padded GEMM M (44*16)
#define CPAD    704            // spat channel stride
#define TOUT    500            // loss time points
#define MAGIC   (400.0f / 750.0f)

#define KC      64             // K-slice per pipeline stage
#define LROW    36             // LDS row stride in dwords (32 data + 4 pad, 144B: 16B aligned)
#define LBUF    (64 * LROW)    // one LDS buffer (dwords)

// ---------------- WMMA vector types ----------------
typedef __attribute__((ext_vector_type(16))) __bf16 v16bf;
typedef __attribute__((ext_vector_type(8)))  float  v8f;

// ---------------- helpers ----------------
// round-to-nearest-even bias in 32-bit integer form
__device__ __forceinline__ unsigned int rne_bias(unsigned int u) {
    return u + 0x7FFFu + ((u >> 16) & 1u);
}
// f32->bf16 RNE pair pack: two 32-bit bias adds + one v_perm_b32
// (dst bytes {uh.b3, uh.b2, ul.b3, ul.b2} -> sel 0x07060302)
__device__ __forceinline__ unsigned int pack_bf2(float lo, float hi) {
    unsigned int ul = rne_bias(__float_as_uint(lo));
    unsigned int uh = rne_bias(__float_as_uint(hi));
    return __builtin_amdgcn_perm(uh, ul, 0x07060302u);
}
__device__ __forceinline__ v16bf make_frag(const unsigned int* lo, const unsigned int* hi) {
    union { unsigned int u[8]; v16bf v; } t;
    uint4 a = *(const uint4*)lo;
    uint4 b = *(const uint4*)hi;
    t.u[0] = a.x; t.u[1] = a.y; t.u[2] = a.z; t.u[3] = a.w;
    t.u[4] = b.x; t.u[5] = b.y; t.u[6] = b.z; t.u[7] = b.w;
    return t.v;
}

// ============================================================
// Kernel 1: jitter + f32->bf16 pack, transposed panel jitT[n][p]
// grid (IMG_H, NPAD), block 128 (2 pixels / thread, 1 dword store)
// ============================================================
__global__ __launch_bounds__(128) void jitter_pack(const float* __restrict__ img,
                                                   const int* __restrict__ eye,
                                                   unsigned short* __restrict__ jitT) {
    const int h  = blockIdx.x;          // 0..159
    const int n  = blockIdx.y;          // 0..511
    const int w0 = threadIdx.x * 2;     // 0..254
    unsigned int packed = 0u;
    if (n < NCOLS) {
        const int b  = n / NJIT;
        const int f  = n - b * NJIT;
        const int dy = eye[(b * NJIT + f) * 2 + 0];
        const int dx = eye[(b * NJIT + f) * 2 + 1];
        const int r  = h + dy;
        if (r < IMG_H) {
            const float* row = img + ((size_t)b * IMG_H + r) * IMG_W;
            float v0 = (w0     + dx < IMG_W) ? row[w0     + dx] : 0.0f;
            float v1 = (w0 + 1 + dx < IMG_W) ? row[w0 + 1 + dx] : 0.0f;
            packed = pack_bf2(v0, v1);
        }
    }
    *(unsigned int*)(jitT + (size_t)n * P_TOT + h * IMG_W + w0) = packed;
}

// ============================================================
// Kernel 1b: copy history frames into spat[b][0..29][c]
// ============================================================
__global__ __launch_bounds__(256) void hist_copy(const float* __restrict__ hist,
                                                 float* __restrict__ spat) {
    const int idx = blockIdx.x * 256 + threadIdx.x;
    if (idx >= BATCH * NHIST * CDIM) return;
    const int b   = idx / (NHIST * CDIM);
    const int rem = idx - b * (NHIST * CDIM);
    const int f   = rem / CDIM;
    const int c   = rem - f * CDIM;
    spat[((size_t)(b * NFRAMES + f)) * CPAD + c] = hist[idx];
}

// ============================================================
// Kernel 2: bf16 WMMA GEMM  spat_j[c,n] = sum_p filt[c,p]*jitT[n,p]
// grid (11, 8): 64M x 64N block tile, 8 waves of 16M x 32N.
// K-step 64, double-buffered LDS, register prefetch of next slice,
// one barrier per 4 WMMAs.
// ============================================================
__global__ __launch_bounds__(256) void gemm_wmma(const float* __restrict__ filt,
                                                 const unsigned short* __restrict__ jitT,
                                                 float* __restrict__ spat) {
    __shared__ unsigned int A_lds[2 * LBUF];   // 2 x (64 rows x 64 bf16 + pad) = 18KB
    __shared__ unsigned int B_lds[2 * LBUF];

    const int tid  = threadIdx.x;
    const int lane = tid & 31;
    const int wave = tid >> 5;
    const int m0   = blockIdx.x * 64;
    const int n0   = blockIdx.y * 64;
    const int mw   = (wave & 3) * 16;     // wave's M sub-tile
    const int nw   = (wave >> 2) * 32;    // wave's N sub-tile (2 frags)

    const int ldRow = tid >> 2;           // 0..63
    const int ldSeg = tid & 3;            // 0..3 (16 elements each)

    // fragment lane patterns (16-bit data, contiguous k = 16*half + s)
    const int frRow = lane & 15;
    const int halfA = (lane < 16) ? 0 : 8;   // A lanes>=16 start at K=16 (dwords)
    const int halfB = (lane < 16) ? 0 : 8;   // B lanes>=16 start at K=16 (dwords)

    v8f acc0 = {};
    v8f acc1 = {};

    const int mA = m0 + ldRow;
    const float*          srcA = filt + (size_t)mA * P_TOT + ldSeg * 16;
    const unsigned short* srcB = jitT + (size_t)(n0 + ldRow) * P_TOT + ldSeg * 16;

    // ---- register-staged slice (software pipeline) ----
    float4 fa0 = {0,0,0,0}, fa1 = {0,0,0,0}, fa2 = {0,0,0,0}, fa3 = {0,0,0,0};
    uint4  rb0 = {0,0,0,0}, rb1 = {0,0,0,0};

    auto load_slice = [&](int k0) {
        if (mA < CDIM) {
            fa0 = *(const float4*)(srcA + k0);
            fa1 = *(const float4*)(srcA + k0 + 4);
            fa2 = *(const float4*)(srcA + k0 + 8);
            fa3 = *(const float4*)(srcA + k0 + 12);
        }
        rb0 = *(const uint4*)(srcB + k0);
        rb1 = *(const uint4*)(srcB + k0 + 8);
    };

    load_slice(0);                        // prologue

    const int NIT = P_TOT / KC;           // 640
    for (int i = 0; i < NIT; ++i) {
        const int buf = (i & 1) * LBUF;

        // ---- pack + store current slice into LDS[buf] ----
        uint4 p0, p1;
        p0.x = pack_bf2(fa0.x, fa0.y);  p0.y = pack_bf2(fa0.z, fa0.w);
        p0.z = pack_bf2(fa1.x, fa1.y);  p0.w = pack_bf2(fa1.z, fa1.w);
        p1.x = pack_bf2(fa2.x, fa2.y);  p1.y = pack_bf2(fa2.z, fa2.w);
        p1.z = pack_bf2(fa3.x, fa3.y);  p1.w = pack_bf2(fa3.z, fa3.w);
        *(uint4*)&A_lds[buf + ldRow * LROW + ldSeg * 8]     = p0;
        *(uint4*)&A_lds[buf + ldRow * LROW + ldSeg * 8 + 4] = p1;
        *(uint4*)&B_lds[buf + ldRow * LROW + ldSeg * 8]     = rb0;
        *(uint4*)&B_lds[buf + ldRow * LROW + ldSeg * 8 + 4] = rb1;

        // ---- issue next slice's global loads (overlap barrier + WMMA) ----
        if (i + 1 < NIT) load_slice((i + 1) * KC);
        // pull slice-after-next of the HBM-heavy A stream into L2
        if (i + 2 < NIT && mA < CDIM) __builtin_prefetch(srcA + (i + 2) * KC, 0, 1);

        __syncthreads();   // LDS[buf] visible; buf^1 free (readers consumed via own dscnt wait)

        // ---- fragments from LDS[buf] (contiguous-in-K b128 loads) ----
        const unsigned int* aRow  = &A_lds[buf + (mw + frRow) * LROW];
        const unsigned int* bRow0 = &B_lds[buf + (nw + frRow) * LROW];
        const unsigned int* bRow1 = &B_lds[buf + (nw + 16 + frRow) * LROW];

        v16bf a0  = make_frag(aRow + halfA,       aRow + halfA + 4);
        v16bf a1  = make_frag(aRow + 16 + halfA,  aRow + 16 + halfA + 4);
        v16bf b00 = make_frag(bRow0 + halfB,      bRow0 + halfB + 4);
        v16bf b01 = make_frag(bRow0 + 16 + halfB, bRow0 + 16 + halfB + 4);
        v16bf b10 = make_frag(bRow1 + halfB,      bRow1 + halfB + 4);
        v16bf b11 = make_frag(bRow1 + 16 + halfB, bRow1 + 16 + halfB + 4);

        // alternate accumulators: no back-to-back RAW on same D registers
        acc0 = __builtin_amdgcn_wmma_f32_16x16x32_bf16(false, a0, false, b00,
                                                       (short)0, acc0, false, false);
        acc1 = __builtin_amdgcn_wmma_f32_16x16x32_bf16(false, a0, false, b10,
                                                       (short)0, acc1, false, false);
        acc0 = __builtin_amdgcn_wmma_f32_16x16x32_bf16(false, a1, false, b01,
                                                       (short)0, acc0, false, false);
        acc1 = __builtin_amdgcn_wmma_f32_16x16x32_bf16(false, a1, false, b11,
                                                       (short)0, acc1, false, false);
    }

    // ---- store C: lane<16 rows 0..7, lane>=16 rows 8..15; col = frRow ----
    const int mbase = m0 + mw + ((lane < 16) ? 0 : 8);
    {
        const int n = n0 + nw + frRow;
        if (n < NCOLS) {
            const int b = n / NJIT, f = n - (n / NJIT) * NJIT;
            float* dst = spat + ((size_t)(b * NFRAMES + NHIST + f)) * CPAD + mbase;
            #pragma unroll
            for (int r = 0; r < 8; ++r)
                if (mbase + r < CDIM) dst[r] = acc0[r];
        }
    }
    {
        const int n = n0 + nw + 16 + frRow;
        if (n < NCOLS) {
            const int b = n / NJIT, f = n - (n / NJIT) * NJIT;
            float* dst = spat + ((size_t)(b * NFRAMES + NHIST + f)) * CPAD + mbase;
            #pragma unroll
            for (int r = 0; r < 8; ++r)
                if (mbase + r < CDIM) dst[r] = acc1[r];
        }
    }
}

// ============================================================
// Kernel 3: per (b,c): gather/upsample + 250-tap corr + loss
// grid (700, 8), 256 threads
// ============================================================
__global__ __launch_bounds__(256) void conv_loss(const float* __restrict__ spat,
                                                 const int* __restrict__ fsel,
                                                 const float* __restrict__ fw,
                                                 const float* __restrict__ tc,
                                                 const float* __restrict__ fb,
                                                 const float* __restrict__ spikes,
                                                 const float* __restrict__ mask,
                                                 float* __restrict__ partial) {
    __shared__ float s_fr[NFRAMES];
    __shared__ float u_lds[NBINS];
    __shared__ float tc_lds[KTAPS];
    __shared__ float red[256];

    const int c = blockIdx.x;
    const int b = blockIdx.y;
    const int tid = threadIdx.x;

    if (tid < NFRAMES) s_fr[tid]   = spat[((size_t)(b * NFRAMES + tid)) * CPAD + c];
    if (tid < KTAPS)   tc_lds[tid] = tc[c * KTAPS + tid];
    __syncthreads();

    for (int t = tid; t < NBINS; t += 256) {
        const int base = (b * NBINS + t) * 2;
        const int s0 = fsel[base + 0];
        const int s1 = fsel[base + 1];
        const float w0 = fw[base + 0];
        const float w1 = fw[base + 1];
        u_lds[t] = s_fr[s0] * w0 + s_fr[s1] * w1;
    }
    __syncthreads();

    float lsum = 0.0f;
    if (tid < 250) {                       // t0 = tid, t1 = tid + 250  (covers 0..499)
        float a0 = 0.0f, a1 = 0.0f;
        #pragma unroll 5
        for (int k = 0; k < KTAPS; ++k) {
            const float tv = tc_lds[k];
            a0 = fmaf(u_lds[tid + k],       tv, a0);
            a1 = fmaf(u_lds[tid + 250 + k], tv, a1);
        }
        const size_t bc = (size_t)b * CDIM + c;
        {
            const float g = a0 + fb[bc * (TOUT + 1) + tid];
            const float s = spikes[bc * NBINS + KTAPS + tid];
            lsum += (__expf(g) - s * g) * mask[b * TOUT + tid];
        }
        {
            const int t1 = tid + 250;
            const float g = a1 + fb[bc * (TOUT + 1) + t1];
            const float s = spikes[bc * NBINS + KTAPS + t1];
            lsum += (__expf(g) - s * g) * mask[b * TOUT + t1];
        }
        lsum *= MAGIC;
    }
    red[tid] = lsum;
    __syncthreads();
    for (int off = 128; off > 0; off >>= 1) {
        if (tid < off) red[tid] += red[tid + off];
        __syncthreads();
    }
    if (tid == 0) partial[b * CDIM + c] = red[0];
}

// ============================================================
// Kernel 4: deterministic per-batch reduction of 700 partials
// ============================================================
__global__ __launch_bounds__(256) void final_reduce(const float* __restrict__ partial,
                                                    float* __restrict__ out) {
    __shared__ float red[256];
    const int b = blockIdx.x;
    const int tid = threadIdx.x;
    float s = 0.0f;
    for (int c = tid; c < CDIM; c += 256) s += partial[b * CDIM + c];
    red[tid] = s;
    __syncthreads();
    for (int off = 128; off > 0; off >>= 1) {
        if (tid < off) red[tid] += red[tid + off];
        __syncthreads();
    }
    if (tid == 0) out[b] = red[0];
}

// ============================================================
extern "C" void kernel_launch(void* const* d_in, const int* in_sizes, int n_in,
                              void* d_out, int out_size, void* d_ws, size_t ws_size,
                              hipStream_t stream) {
    (void)in_sizes; (void)n_in; (void)out_size; (void)ws_size;

    const float* img    = (const float*)d_in[0];  // (B,H,W)
    const float* spikes = (const float*)d_in[1];  // (B,C,750)
    const int*   eye    = (const int*)  d_in[2];  // (B,60,2)
    const float* mask   = (const float*)d_in[3];  // (B,500)
    const int*   fsel   = (const int*)  d_in[4];  // (B,750,2)
    const float* fw     = (const float*)d_in[5];  // (B,750,2)
    const float* filt   = (const float*)d_in[6];  // (C,40960)
    const float* tc     = (const float*)d_in[7];  // (C,250)
    const float* fb     = (const float*)d_in[8];  // (B,C,501)
    const float* hist   = (const float*)d_in[9];  // (B,30,C)
    float* out = (float*)d_out;                   // (B,) f32

    char* ws = (char*)d_ws;
    const size_t jitT_bytes = (size_t)NPAD * P_TOT * sizeof(unsigned short);  // 41,943,040
    const size_t spat_bytes = (size_t)BATCH * NFRAMES * CPAD * sizeof(float); //  2,027,520
    unsigned short* jitT    = (unsigned short*)ws;
    float*          spat    = (float*)(ws + jitT_bytes);
    float*          partial = (float*)(ws + jitT_bytes + spat_bytes);

    jitter_pack<<<dim3(IMG_H, NPAD), dim3(128), 0, stream>>>(img, eye, jitT);
    hist_copy<<<dim3((BATCH * NHIST * CDIM + 255) / 256), dim3(256), 0, stream>>>(hist, spat);
    gemm_wmma<<<dim3(MPAD / 64, NPAD / 64), dim3(256), 0, stream>>>(filt, jitT, spat);
    conv_loss<<<dim3(CDIM, BATCH), dim3(256), 0, stream>>>(spat, fsel, fw, tc, fb,
                                                           spikes, mask, partial);
    final_reduce<<<dim3(BATCH), dim3(256), 0, stream>>>(partial, out);
}
